// MultiHeadAttention_33844342292657
// MI455X (gfx1250) — matrix operand order repaired and checked by reading, not compiled, not run on hardware
//
#include <hip/hip_runtime.h>

// Problem constants (match reference)
constexpr int cB  = 2;
constexpr int cS  = 2048;
constexpr int cD  = 1024;
constexpr int cH  = 16;
constexpr int cDK = 64;
constexpr int cM  = cB * cS;      // 4096 rows for the projection GEMMs

typedef __attribute__((ext_vector_type(16))) __bf16 bf16x16;
typedef __attribute__((ext_vector_type(8)))  __bf16 bf16x8;
typedef __attribute__((ext_vector_type(4)))  __bf16 bf16x4;
typedef __attribute__((ext_vector_type(8)))  float  f32x8;
typedef __attribute__((ext_vector_type(4)))  unsigned int u32x4;
typedef __attribute__((ext_vector_type(8)))  int i32x8;
typedef __attribute__((ext_vector_type(4)))  int i32x4;

__device__ __forceinline__ f32x8 wmma_bf16(bf16x16 a, bf16x16 b, f32x8 c) {
  return __builtin_amdgcn_wmma_f32_16x16x32_bf16(
      /*neg_a=*/false, a, /*neg_b=*/false, b,
      /*c_mod=*/(short)0, c, /*reuse_a=*/false, /*reuse_b=*/false);
}

// A-operand (16x32, 16-bit): lane m (lo half) holds K = {0..15},
// lane m (hi half) holds K = {16..31}. One contiguous 16-element chunk.
__device__ __forceinline__ bf16x16 load_a16x32(const __bf16* base, int half) {
  bf16x8 lo = *(const bf16x8*)(base + half * 16);
  bf16x8 hi = *(const bf16x8*)(base + half * 16 + 8);
  return __builtin_shufflevector(lo, hi, 0, 1, 2, 3, 4, 5, 6, 7,
                                         8, 9, 10, 11, 12, 13, 14, 15);
}

// ---------------------------------------------------------------------------
// Tensor Data Mover: 2D tile (rows x 64 bf16 cols) from global -> LDS.
// D# per CDNA5 ISA. data_size=1 (2 bytes).
// This toolchain exposes the 6-arg builtin: (g0, g1, g2, g3, g_extra, cpol).
// ---------------------------------------------------------------------------
__device__ __forceinline__ void tdm_load_2d(const void* lds_dst,
                                            const void* gsrc,
                                            unsigned tensor_d0, unsigned tensor_d1,
                                            unsigned tile_d0,   unsigned tile_d1,
                                            unsigned row_stride_elems) {
  unsigned long long ga = (unsigned long long)(size_t)gsrc;
  unsigned lds_off = (unsigned)(size_t)lds_dst;   // LDS aperture: addr[31:0] = offset
  u32x4 g0;
  g0[0] = 1u;                                     // count=1, user descriptor
  g0[1] = lds_off;                                // lds_addr [63:32]
  g0[2] = (unsigned)(ga & 0xffffffffu);           // global_addr lo
  g0[3] = (unsigned)((ga >> 32) & 0x1ffffffu)     // global_addr hi (57-bit)
          | (2u << 30);                           // type=2 ("image")
  i32x8 g1;
  g1[0] = (int)(1u << 16);                        // workgroup_mask=0, data_size=1(2B)
  g1[1] = (int)((tensor_d0 & 0xffffu) << 16);     // tensor_dim0 low16 @ [63:48]
  g1[2] = (int)((tensor_d0 >> 16) | ((tensor_d1 & 0xffffu) << 16)); // d0 hi | d1 lo
  g1[3] = (int)((tensor_d1 >> 16) | (tile_d0 << 16));               // d1 hi | tile_dim0
  g1[4] = (int)(tile_d1 & 0xffffu);               // tile_dim1 | tile_dim2=0
  g1[5] = (int)row_stride_elems;                  // tensor_dim0_stride low32
  g1[6] = 0;                                      // stride hi | dim1_stride lo (2D: unused)
  g1[7] = 0;
  i32x4 z4 = {0, 0, 0, 0};                        // groups 2/3 unused (<=2D)
  i32x8 z8 = {0, 0, 0, 0, 0, 0, 0, 0};
  __builtin_amdgcn_tensor_load_to_lds(g0, g1, z4, z4, z8, 0);
}

// ---------------------------------------------------------------------------
// Kernel 1: f32 -> bf16 elementwise conversion (vectorized x4)
// ---------------------------------------------------------------------------
__global__ __launch_bounds__(256)
void cvt_f32_bf16_kernel(const float* __restrict__ src,
                         __bf16* __restrict__ dst, int n4) {
  int i = blockIdx.x * blockDim.x + threadIdx.x;
  if (i < n4) {
    float4 f = ((const float4*)src)[i];
    bf16x4 o = { (__bf16)f.x, (__bf16)f.y, (__bf16)f.z, (__bf16)f.w };
    ((bf16x4*)dst)[i] = o;
  }
}

// ---------------------------------------------------------------------------
// Kernel 2/4: C[M,N] = A[M,K] @ W[N,K]^T   (bf16 in, WMMA f32 accumulate)
//   mode 0: scatter bf16 result into per-head layout [B,H,S,DK]
//   mode 1: f32 result row-major [M,N] (final output projection)
// Block 128 (4 waves). Workgroup tile 128(M) x 64(N); each wave owns 32x64
// (2 M-subtiles share the 4 B operands -> halved LDS traffic per WMMA).
// A/W tiles DMA'd by the Tensor Data Mover into double-buffered LDS; the
// next tile's TDM is in flight while the current one is consumed.
// ---------------------------------------------------------------------------
__global__ __launch_bounds__(128)
void gemm_bf16_wmma(const __bf16* __restrict__ A,
                    const __bf16* __restrict__ W,
                    __bf16* __restrict__ OutHead,
                    float* __restrict__ OutF32,
                    int mode) {
  __shared__ __bf16 As[2][128 * 64] __attribute__((aligned(64)));
  __shared__ __bf16 Ws[2][64 * 64] __attribute__((aligned(64)));

  const int tid  = threadIdx.x;
  const int lane = tid & 31;
  const int wave = tid >> 5;
  const int ln   = lane & 15;
  const int half = lane >> 4;
  const int n0   = blockIdx.x * 64;
  const int m0   = blockIdx.y * 128;

  f32x8 acc[2][4] = {};

  const __bf16* Ablk = A + (size_t)m0 * cD;
  const __bf16* Wblk = W + (size_t)n0 * cD;
  const int nk = cD / 64;   // 16 K-steps of 64

  // Prologue: DMA first tiles (wave0 -> A tile, wave1 -> W tile).
  if (wave == 0) tdm_load_2d(&As[0][0], Ablk, cD, 128, 64, 128, cD);
  if (wave == 1) tdm_load_2d(&Ws[0][0], Wblk, cD, 64, 64, 64, cD);
  if (wave <= 1) __builtin_amdgcn_s_wait_tensorcnt(0);
  __syncthreads();

  for (int ki = 0; ki < nk; ++ki) {
    const int cur = ki & 1, nxt = cur ^ 1;
    if (ki + 1 < nk) {   // kick off next tile's DMA before computing this one
      const int kk = (ki + 1) * 64;
      if (wave == 0) tdm_load_2d(&As[nxt][0], Ablk + kk, cD, 128, 64, 128, cD);
      if (wave == 1) tdm_load_2d(&Ws[nxt][0], Wblk + kk, cD, 64, 64, 64, cD);
    }

    const __bf16* as = &As[cur][0];
    const __bf16* ws = &Ws[cur][0];
#pragma unroll
    for (int ks = 0; ks < 2; ++ks) {
      // Register-block: preload all operands of this k-step, then 8 WMMAs.
      bf16x16 a0 = load_a16x32(as + (wave * 32 + ln) * 64 + ks * 32, half);
      bf16x16 a1 = load_a16x32(as + (wave * 32 + 16 + ln) * 64 + ks * 32, half);
      bf16x16 bw[4];
#pragma unroll
      for (int ns = 0; ns < 4; ++ns)
        bw[ns] = *(const bf16x16*)(ws + (ns * 16 + ln) * 64 + ks * 32 + half * 16);
#pragma unroll
      for (int ns = 0; ns < 4; ++ns) acc[0][ns] = wmma_bf16(a0, bw[ns], acc[0][ns]);
#pragma unroll
      for (int ns = 0; ns < 4; ++ns) acc[1][ns] = wmma_bf16(a1, bw[ns], acc[1][ns]);
    }

    if (wave <= 1 && ki + 1 < nk) __builtin_amdgcn_s_wait_tensorcnt(0);
    __syncthreads();
  }

  // Epilogue. C layout: VGPR r, lanes 0-15 -> (M=r, N=ln); lanes 16-31 -> M=r+8.
  const int mrow0 = m0 + wave * 32;
#pragma unroll
  for (int msub = 0; msub < 2; ++msub) {
#pragma unroll
    for (int ns = 0; ns < 4; ++ns) {
#pragma unroll
      for (int r = 0; r < 8; ++r) {
        int mg = mrow0 + msub * 16 + r + half * 8;
        int ng = n0 + ns * 16 + ln;
        float v = acc[msub][ns][r];
        if (mode == 0) {
          int b  = mg / cS;
          int s  = mg & (cS - 1);
          int h  = ng / cDK;
          int dk = ng & (cDK - 1);
          OutHead[(((size_t)(b * cH + h) * cS + s) * cDK) + dk] = (__bf16)v;
        } else {
          OutF32[(size_t)mg * cD + ng] = v;
        }
      }
    }
  }
}

// ---------------------------------------------------------------------------
// Kernel 3: causal flash attention over per-head bf16 Q/K/V [B,H,S,DK].
// Grid: (S/64, B*H). Block: 128 (4 waves); wave handles 16 query rows.
// K tile arrives via TDM while all threads transpose the V tile into LDS.
// ---------------------------------------------------------------------------
__global__ __launch_bounds__(128)
void flash_attn_wmma(const __bf16* __restrict__ Qh,
                     const __bf16* __restrict__ Kh,
                     const __bf16* __restrict__ Vh,
                     __bf16* __restrict__ Out,
                     float scale) {
  __shared__ __bf16 Ks[64 * 64] __attribute__((aligned(64)));     // [kpos][dk] (TDM)
  __shared__ __bf16 Vt[64 * 64] __attribute__((aligned(64)));     // [dk][kpos]
  __shared__ __bf16 Ps[4 * 16 * 64] __attribute__((aligned(64))); // per-wave P

  const int tid  = threadIdx.x;
  const int lane = tid & 31;
  const int wave = tid >> 5;
  const int ln   = lane & 15;
  const int half = lane >> 4;
  const int q0   = blockIdx.x * 64;
  const int bh   = blockIdx.y;                 // b*H + h
  const int b    = bh / cH;
  const int h    = bh & (cH - 1);

  const __bf16* Qb = Qh + (size_t)bh * cS * cDK;
  const __bf16* Kb = Kh + (size_t)bh * cS * cDK;
  const __bf16* Vb = Vh + (size_t)bh * cS * cDK;

  // Preload this wave's Q operands (rows q0 + wave*16 + [0..15], DK=64 -> 2 k-steps)
  const int qrow = q0 + wave * 16 + ln;
  bf16x16 aq0 = load_a16x32(Qb + (size_t)qrow * cDK + 0,  half);
  bf16x16 aq1 = load_a16x32(Qb + (size_t)qrow * cDK + 32, half);

  f32x8 acc[4] = {};
  float mrow[8], lrow[8];
#pragma unroll
  for (int r = 0; r < 8; ++r) { mrow[r] = -3.0e38f; lrow[r] = 0.0f; }

  const int ntiles = q0 / 64 + 1;              // causal: keys <= q0+63
  for (int kt = 0; kt < ntiles; ++kt) {
    const int kbase = kt * 64;

    // K tile: DMA via Tensor Data Mover (overlaps the V transpose below).
    if (wave == 0)
      tdm_load_2d(&Ks[0], Kb + (size_t)kbase * cDK, cDK, 64, 64, 64, cDK);

    // V tile: cooperative load + transpose into LDS (all 128 threads).
    for (int c = tid; c < 512; c += 128) {
      int r = c >> 3, cc = c & 7;
      bf16x8 v = *(const bf16x8*)(Vb + (size_t)(kbase + r) * cDK + cc * 8);
#pragma unroll
      for (int j = 0; j < 8; ++j) Vt[(cc * 8 + j) * 64 + r] = v[j];
    }
    if (kt + 1 < ntiles) {       // pull next V tile toward L2 (global_prefetch_b8)
      __builtin_prefetch(Vb + (size_t)(kbase + 64) * cDK + tid * 32, 0, 1);
    }
    if (wave == 0) __builtin_amdgcn_s_wait_tensorcnt(0);
    __syncthreads();

    // scores S = Q @ K^T: preload 4 B operands per k-step, then WMMA chain.
    f32x8 s[4] = {};
#pragma unroll
    for (int ks = 0; ks < 2; ++ks) {
      bf16x16 bk[4];
#pragma unroll
      for (int ns = 0; ns < 4; ++ns)
        bk[ns] = *(const bf16x16*)(Ks + (ns * 16 + ln) * 64 + ks * 32 + half * 16);
      bf16x16 a = (ks == 0) ? aq0 : aq1;
#pragma unroll
      for (int ns = 0; ns < 4; ++ns) s[ns] = wmma_bf16(a, bk[ns], s[ns]);
    }

    // scale + causal mask
#pragma unroll
    for (int ns = 0; ns < 4; ++ns) {
      int kg = kbase + ns * 16 + ln;
#pragma unroll
      for (int r = 0; r < 8; ++r) {
        int qg = q0 + wave * 16 + r + half * 8;
        float v = s[ns][r] * scale;
        s[ns][r] = (kg > qg) ? -1.0e9f : v;
      }
    }

    // online softmax update (row stats live per 16-lane half)
#pragma unroll
    for (int r = 0; r < 8; ++r) {
      float tmax = fmaxf(fmaxf(s[0][r], s[1][r]), fmaxf(s[2][r], s[3][r]));
#pragma unroll
      for (int m = 1; m <= 8; m <<= 1)
        tmax = fmaxf(tmax, __shfl_xor(tmax, m, 32));
      float mnew = fmaxf(mrow[r], tmax);
      float esc  = __expf(mrow[r] - mnew);
      mrow[r] = mnew;
      lrow[r] *= esc;
#pragma unroll
      for (int ns = 0; ns < 4; ++ns) acc[ns][r] *= esc;
      float rs = 0.0f;
#pragma unroll
      for (int ns = 0; ns < 4; ++ns) {
        float p = __expf(s[ns][r] - mnew);
        s[ns][r] = p;
        rs += p;
      }
#pragma unroll
      for (int m = 1; m <= 8; m <<= 1)
        rs += __shfl_xor(rs, m, 32);
      lrow[r] += rs;
    }

    // Re-layout P: C-layout -> LDS row-major -> A-layout (per-wave region).
    __bf16* Pw = Ps + wave * 16 * 64;
#pragma unroll
    for (int ns = 0; ns < 4; ++ns)
#pragma unroll
      for (int r = 0; r < 8; ++r)
        Pw[(r + half * 8) * 64 + ns * 16 + ln] = (__bf16)s[ns][r];

    // O += P @ V  (V transposed in LDS -> contiguous B operand)
#pragma unroll
    for (int ks = 0; ks < 2; ++ks) {
      bf16x16 ap = load_a16x32(Pw + ln * 64 + ks * 32, half);
      bf16x16 bv[4];
#pragma unroll
      for (int ns = 0; ns < 4; ++ns)
        bv[ns] = *(const bf16x16*)(Vt + (ns * 16 + ln) * 64 + ks * 32 + half * 16);
#pragma unroll
      for (int ns = 0; ns < 4; ++ns) acc[ns] = wmma_bf16(ap, bv[ns], acc[ns]);
    }
    __syncthreads();
  }

  // Epilogue: merged-head output [B, S, H*DK]
#pragma unroll
  for (int r = 0; r < 8; ++r) {
    float inv = 1.0f / lrow[r];
    int qg = q0 + wave * 16 + r + half * 8;
    size_t rowoff = ((size_t)b * cS + qg) * (size_t)cD + (size_t)h * cDK;
#pragma unroll
    for (int ns = 0; ns < 4; ++ns)
      Out[rowoff + ns * 16 + ln] = (__bf16)(acc[ns][r] * inv);
  }
}

// ---------------------------------------------------------------------------
// Host-side orchestration
// ---------------------------------------------------------------------------
extern "C" void kernel_launch(void* const* d_in, const int* in_sizes, int n_in,
                              void* d_out, int out_size, void* d_ws, size_t ws_size,
                              hipStream_t stream) {
  const float* q  = (const float*)d_in[0];
  const float* k  = (const float*)d_in[1];
  const float* v  = (const float*)d_in[2];
  // d_in[3] = causal mask (recomputed analytically in-kernel)
  const float* Wq = (const float*)d_in[4];
  const float* Wk = (const float*)d_in[5];
  const float* Wv = (const float*)d_in[6];
  const float* Wo = (const float*)d_in[7];
  float* out = (float*)d_out;

  const size_t nX = (size_t)cM * cD;   // 4,194,304
  const size_t nW = (size_t)cD * cD;   // 1,048,576

  __bf16* ws  = (__bf16*)d_ws;
  __bf16* xq  = ws;                    // bf16 activations
  __bf16* xk  = xq + nX;
  __bf16* xv  = xk + nX;
  __bf16* wq  = xv + nX;               // bf16 weights
  __bf16* wk  = wq + nW;
  __bf16* wv  = wk + nW;
  __bf16* wo  = wv + nW;
  __bf16* Qh  = wo + nW;               // per-head projected [B,H,S,DK]
  __bf16* Kh  = Qh + nX;
  __bf16* Vh  = Kh + nX;
  __bf16* att = Vh + nX;               // merged-head attention out [B,S,D]
  const size_t needed = ((size_t)(att + nX) - (size_t)d_ws);
  if (ws_size < needed) return;        // deterministic guard

  // 1) f32 -> bf16 conversions
  cvt_f32_bf16_kernel<<<(int)(nX / 4 / 256), 256, 0, stream>>>(q,  xq, (int)(nX / 4));
  cvt_f32_bf16_kernel<<<(int)(nX / 4 / 256), 256, 0, stream>>>(k,  xk, (int)(nX / 4));
  cvt_f32_bf16_kernel<<<(int)(nX / 4 / 256), 256, 0, stream>>>(v,  xv, (int)(nX / 4));
  cvt_f32_bf16_kernel<<<(int)(nW / 4 / 256), 256, 0, stream>>>(Wq, wq, (int)(nW / 4));
  cvt_f32_bf16_kernel<<<(int)(nW / 4 / 256), 256, 0, stream>>>(Wk, wk, (int)(nW / 4));
  cvt_f32_bf16_kernel<<<(int)(nW / 4 / 256), 256, 0, stream>>>(Wv, wv, (int)(nW / 4));
  cvt_f32_bf16_kernel<<<(int)(nW / 4 / 256), 256, 0, stream>>>(Wo, wo, (int)(nW / 4));

  // 2) Q/K/V projections: y = x @ W^T, scattered into per-head layout
  dim3 gg(cD / 64, cM / 128);          // (16, 32)
  gemm_bf16_wmma<<<gg, 128, 0, stream>>>(xq, wq, Qh, nullptr, 0);
  gemm_bf16_wmma<<<gg, 128, 0, stream>>>(xk, wk, Kh, nullptr, 0);
  gemm_bf16_wmma<<<gg, 128, 0, stream>>>(xv, wv, Vh, nullptr, 0);

  // 3) causal flash attention
  dim3 ga(cS / 64, cB * cH);           // (32, 32)
  flash_attn_wmma<<<ga, 128, 0, stream>>>(Qh, Kh, Vh, att, 0.125f /* 1/sqrt(64) */);

  // 4) output projection, f32 straight into d_out
  gemm_bf16_wmma<<<gg, 128, 0, stream>>>(att, wo, nullptr, out, 1);
}
